// LTITRFGen_4629974745720
// MI455X (gfx1250) — matrix operand order repaired and checked by reading, not compiled, not run on hardware
//
#include <hip/hip_runtime.h>
#include <hip/hip_bf16.h>

// Problem constants: x:(128,8192) f32, W:(128,64,128) f32, bias:(128,)
// out[n, w, o] = sum_i x[i,n]*W[i,w,o] + 128*bias[o]
// GEMM view: C[M=8192 (n), N=8192 (wo=w*128+o)] = X^T * W, K=128.
#define IN_DIM   128
#define NCOLS    8192          // n dimension (M)
#define WODIM    8192          // wo dimension (N) = 64*128
#define KB_COUNT 4             // K blocks of 32 (bf16 WMMA depth)

typedef __attribute__((ext_vector_type(16))) __bf16 v16bf_t;
typedef __attribute__((ext_vector_type(8)))  float  v8f_t;

// ---------------------------------------------------------------------------
// RNE f32 -> bf16, and hi/lo split:  f ~= hi + lo  (each bf16)
// ---------------------------------------------------------------------------
__device__ __forceinline__ unsigned short f32_to_bf16_rne(float f) {
  unsigned u = __float_as_uint(f);
  unsigned r = u + 0x7FFFu + ((u >> 16) & 1u);
  return (unsigned short)(r >> 16);
}
__device__ __forceinline__ void split_bf16(float f, unsigned short& hi, unsigned short& lo) {
  hi = f32_to_bf16_rne(f);
  float hif = __uint_as_float((unsigned)hi << 16);
  lo = f32_to_bf16_rne(f - hif);
}

// ---------------------------------------------------------------------------
// Pack X^T into A-fragment layout (16-bit A 16x32, CDNA5 ISA §7.12.2):
//   lane L: M = L%16, h = L/16; element e (0..15): K_local = (e&7) + 8h + (e>=8 ? 16 : 0)
// Fragment f = row_tile*4 + kb; 512 bf16 per fragment; lane stride 16 elems (32B).
// ---------------------------------------------------------------------------
__global__ void __launch_bounds__(256)
pack_a_kernel(const float* __restrict__ x, unsigned short* __restrict__ a_hi,
              unsigned short* __restrict__ a_lo) {
  int tid  = threadIdx.x;
  int lane = tid & 31;
  int frag = blockIdx.x * 8 + (tid >> 5);   // 512 row tiles * 4 kb = 2048 frags
  int r    = frag >> 2;                     // row tile (16 n's)
  int kb   = frag & 3;
  int h    = lane >> 4;
  int m    = lane & 15;
  int n    = r * 16 + m;
  size_t base = (size_t)frag * 512 + (size_t)lane * 16;
#pragma unroll
  for (int e = 0; e < 16; ++e) {
    int k_local = (e & 7) + 8 * h + ((e >> 3) << 4);
    float v = x[(size_t)(kb * 32 + k_local) * NCOLS + n];
    unsigned short hi, lo;
    split_bf16(v, hi, lo);
    a_hi[base + e] = hi;
    a_lo[base + e] = lo;
  }
}

// ---------------------------------------------------------------------------
// Pack W into B-fragment layout (16-bit B 32x16):
//   lane L: N = L%16, h = L/16; element e: K_local = (e&7) + 8h + (e>=8 ? 16 : 0)
// Fragment g = col_tile*4 + kb.
// ---------------------------------------------------------------------------
__global__ void __launch_bounds__(256)
pack_b_kernel(const float* __restrict__ w, unsigned short* __restrict__ b_hi,
              unsigned short* __restrict__ b_lo) {
  int tid  = threadIdx.x;
  int lane = tid & 31;
  int frag = blockIdx.x * 8 + (tid >> 5);   // 512 col tiles * 4 kb = 2048 frags
  int c    = frag >> 2;                     // col tile (16 wo's)
  int kb   = frag & 3;
  int h    = lane >> 4;
  int m    = lane & 15;
  int wo   = c * 16 + m;
  size_t base = (size_t)frag * 512 + (size_t)lane * 16;
#pragma unroll
  for (int e = 0; e < 16; ++e) {
    int k_local = (e & 7) + 8 * h + ((e >> 3) << 4);
    float v = w[(size_t)(kb * 32 + k_local) * WODIM + wo];
    unsigned short hi, lo;
    split_bf16(v, hi, lo);
    b_hi[base + e] = hi;
    b_lo[base + e] = lo;
  }
}

// ---------------------------------------------------------------------------
// Main GEMM: 8 waves/block; each wave computes a 32x64 C tile.
// Block tile = 128x128. Grid = (8192/128, 8192/128) = (64, 64).
// Split-bf16: acc += Ahi*Bhi + Ahi*Blo + Alo*Bhi  (3 WMMAs per 16x16x32 step).
// ---------------------------------------------------------------------------
__global__ void __launch_bounds__(256)
gemm_bf16x3_kernel(const unsigned short* __restrict__ a_hi,
                   const unsigned short* __restrict__ a_lo,
                   const unsigned short* __restrict__ b_hi,
                   const unsigned short* __restrict__ b_lo,
                   const float* __restrict__ bias,
                   float* __restrict__ out) {
  int tid  = threadIdx.x;
  int lane = tid & 31;
  int wave = tid >> 5;          // 0..7
  int wr   = wave & 3;          // 4 row-waves
  int wc   = wave >> 2;         // 2 col-waves
  int m    = lane & 15;
  int h    = lane >> 4;

  int row_tile0 = blockIdx.y * 8 + wr * 2;  // 16-row units, 2 per wave
  int col_tile0 = blockIdx.x * 8 + wc * 4;  // 16-col units, 4 per wave

  v8f_t acc[2][4] = {};  // zero-initialized accumulators

#pragma unroll
  for (int kb = 0; kb < KB_COUNT; ++kb) {
    v16bf_t ahi[2], alo[2];
#pragma unroll
    for (int rt = 0; rt < 2; ++rt) {
      size_t fa = ((size_t)(row_tile0 + rt) * 4 + kb) * 512 + (size_t)lane * 16;
      ahi[rt] = *(const v16bf_t*)(a_hi + fa);
      alo[rt] = *(const v16bf_t*)(a_lo + fa);
    }
#pragma unroll
    for (int ct = 0; ct < 4; ++ct) {
      size_t fb = ((size_t)(col_tile0 + ct) * 4 + kb) * 512 + (size_t)lane * 16;
      v16bf_t bhi = *(const v16bf_t*)(b_hi + fb);
      v16bf_t blo = *(const v16bf_t*)(b_lo + fb);
#pragma unroll
      for (int rt = 0; rt < 2; ++rt) {
        acc[rt][ct] = __builtin_amdgcn_wmma_f32_16x16x32_bf16(
            false, ahi[rt], false, bhi, (short)0, acc[rt][ct], false, false);
        acc[rt][ct] = __builtin_amdgcn_wmma_f32_16x16x32_bf16(
            false, ahi[rt], false, blo, (short)0, acc[rt][ct], false, false);
        acc[rt][ct] = __builtin_amdgcn_wmma_f32_16x16x32_bf16(
            false, alo[rt], false, bhi, (short)0, acc[rt][ct], false, false);
      }
    }
  }

  // Epilogue: C layout (ISA): VGPR v -> row M = v + 8h, col N = m.
  // out[n*8192 + wo] with bias term 128*bias[wo & 127]. Nontemporal: 256MB
  // output stream, no reuse.
#pragma unroll
  for (int ct = 0; ct < 4; ++ct) {
    int wo = (col_tile0 + ct) * 16 + m;
    float bv = 128.0f * bias[wo & (IN_DIM - 1)];
#pragma unroll
    for (int rt = 0; rt < 2; ++rt) {
      int nb = (row_tile0 + rt) * 16 + 8 * h;
#pragma unroll
      for (int v = 0; v < 8; ++v) {
        __builtin_nontemporal_store(acc[rt][ct][v] + bv,
                                    out + (size_t)(nb + v) * WODIM + wo);
      }
    }
  }
}

// ---------------------------------------------------------------------------
// Deterministic VALU fallback (only if workspace is too small for planes).
// ---------------------------------------------------------------------------
__global__ void __launch_bounds__(256)
fallback_kernel(const float* __restrict__ x, const float* __restrict__ w,
                const float* __restrict__ bias, float* __restrict__ out) {
  size_t idx = (size_t)blockIdx.x * blockDim.x + threadIdx.x;
  size_t n  = idx >> 13;
  size_t wo = idx & (WODIM - 1);
  float s = 0.0f;
#pragma unroll 8
  for (int i = 0; i < IN_DIM; ++i)
    s += x[(size_t)i * NCOLS + n] * w[(size_t)i * WODIM + wo];
  out[idx] = s + 128.0f * bias[wo & (IN_DIM - 1)];
}

extern "C" void kernel_launch(void* const* d_in, const int* in_sizes, int n_in,
                              void* d_out, int out_size, void* d_ws, size_t ws_size,
                              hipStream_t stream) {
  const float* x    = (const float*)d_in[0];  // (128, 8192)
  const float* wgt  = (const float*)d_in[1];  // (128, 64, 128) -> (128, 8192)
  const float* bias = (const float*)d_in[2];  // (128,)
  float* out = (float*)d_out;                 // (8192, 64, 128) -> (8192, 8192)

  const size_t PLANE = (size_t)2048 * 512;    // bf16 elements per plane (2 MB)
  const size_t NEED  = 4 * PLANE * sizeof(unsigned short);  // 8 MB

  if (ws_size >= NEED) {
    unsigned short* a_hi = (unsigned short*)d_ws;
    unsigned short* a_lo = a_hi + PLANE;
    unsigned short* b_hi = a_lo + PLANE;
    unsigned short* b_lo = b_hi + PLANE;

    pack_a_kernel<<<256, 256, 0, stream>>>(x, a_hi, a_lo);
    pack_b_kernel<<<256, 256, 0, stream>>>(wgt, b_hi, b_lo);
    gemm_bf16x3_kernel<<<dim3(64, 64), 256, 0, stream>>>(a_hi, a_lo, b_hi, b_lo,
                                                         bias, out);
  } else {
    size_t total = (size_t)NCOLS * WODIM;
    fallback_kernel<<<(unsigned)(total / 256), 256, 0, stream>>>(x, wgt, bias, out);
  }
}
